// LocalGOCorrOpt_70007966924950
// MI455X (gfx1250) — compile-verified, matrix operand order checked by reading, not run on hardware
//
#include <hip/hip_runtime.h>
#include <hip/hip_bf16.h>

typedef __attribute__((ext_vector_type(16))) __bf16 v16bf;
typedef __attribute__((ext_vector_type(8)))  float  v8f;
typedef __attribute__((ext_vector_type(4)))  unsigned int v4u;
typedef __attribute__((ext_vector_type(8)))  int v8i;
typedef __attribute__((ext_vector_type(4)))  int v4i;

#define NBINS 10
#define NDISP 81
#define HP 72            // padded height (64 + 2*4)
#define WP 80            // padded width  (64 + 4 left + 12 right)

// --------------------------------------------------------------------------
// LDS byte offset of a __shared__ object: flat LDS addresses carry the
// aperture in [63:32] and the LDS offset in [31:0] (ISA 10.2), so truncation
// yields the wave-relative LDS byte address TDM/async ops expect.
// --------------------------------------------------------------------------
__device__ __forceinline__ unsigned int lds_off32(const void* p) {
  return (unsigned int)(unsigned long long)p;
}

// --------------------------------------------------------------------------
// 1-D TDM tile load: nelem 2-byte elements, contiguous in memory, written to
// LDS with 16B of padding inserted after every 512B (bank-conflict padding
// for the 264-element rows). D# per CDNA5 ISA ch.8. This toolchain declares
// the 6-arg builtin: (v4u g0, v8i g1, v4i g2, v4i g3, v8i, i32 cpol).
// --------------------------------------------------------------------------
__device__ __forceinline__ void tdm_load_1d(unsigned int lds_addr,
                                            unsigned long long gaddr,
                                            unsigned int nelem) {
#if __has_builtin(__builtin_amdgcn_tensor_load_to_lds)
  v4u g0;
  g0.x = 1u;                                               // count=1 (valid)
  g0.y = lds_addr;                                         // lds_addr
  g0.z = (unsigned int)gaddr;                              // global_addr[31:0]
  g0.w = (unsigned int)((gaddr >> 32) & 0x1FFFFFFu) | 0x80000000u; // ga[56:32] | type=2
  v8i g1;
  g1[0] = (1 << 16)      // data_size = 1 -> 2 bytes
        | (1 << 20)      // pad_enable
        | (6 << 22)      // pad_interval: 128 DWORDs = 512B
        | (3 << 25);     // pad_amount:   4 DWORDs = 16B
  g1[1] = (int)(nelem << 16);                    // tensor_dim0[15:0]  @ [63:48]
  g1[2] = (int)((nelem >> 16) | (1u << 16));     // tensor_dim0[31:16], tensor_dim1=1
  g1[3] = (int)(nelem << 16);                    // tile_dim0 @ [127:112]
  g1[4] = 1;                                     // tile_dim1 = 1
  g1[5] = (int)nelem;                            // tensor_dim0_stride[31:0]
  g1[6] = 0;
  g1[7] = 0;
  v4i gz4 = {0, 0, 0, 0};
  v8i gz8 = {0, 0, 0, 0, 0, 0, 0, 0};
  __builtin_amdgcn_tensor_load_to_lds(g0, g1, gz4, gz4, gz8, 0);
#else
  (void)lds_addr; (void)gaddr; (void)nelem;
#endif
}

__device__ __forceinline__ void async_ld_b128(unsigned int lds_addr,
                                              unsigned long long gaddr) {
  asm volatile("global_load_async_to_lds_b128 %0, %1, off"
               :: "v"(lds_addr), "v"(gaddr) : "memory");
}
__device__ __forceinline__ void wait_async0() {
  asm volatile("s_wait_asynccnt 0" ::: "memory");
}

// --------------------------------------------------------------------------
// Prep: per-displacement constants into ws header.
// hdr[0..80]=v_plus*target, hdr[81..161]=c0, hdr[162..242]=c1,
// hdr[243]=reg_weight, hdr[244]=step_length
// --------------------------------------------------------------------------
__global__ void prep_kernel(const float* __restrict__ log_step,
                            const float* __restrict__ freg,
                            const float* __restrict__ label_w,
                            const float* __restrict__ spatial_w,
                            const float* __restrict__ mask_w,
                            float* __restrict__ hdr) {
  int d = threadIdx.x;
  if (d < NDISP) {
    int dy = d / 9, dx = d % 9;
    float fy = (float)(dy - 4), fx = (float)(dx - 4);
    float dist = sqrtf(fy * fy + fx * fx);
    float tgt = 0.f, vp = 0.f, ml = 0.f;
    #pragma unroll
    for (int k = 0; k < NBINS; ++k) {
      float bd = dist / 0.5f - (float)k;
      float bin = (k < NBINS - 1) ? fmaxf(0.f, 1.f - fabsf(bd))
                                  : fminf(fmaxf(1.f + bd, 0.f), 1.f);
      tgt += label_w[k] * bin;
      vp  += spatial_w[k] * bin;
      ml  += mask_w[k] * bin;
    }
    float w = 1.f / (1.f + expf(-ml));
    hdr[d]       = vp * tgt;
    hdr[81 + d]  = 0.5f * vp * (1.f - w);
    hdr[162 + d] = 0.5f * vp * (1.f + w);
  }
  if (threadIdx.x == 0) {
    float fr = freg[0];
    hdr[243] = fmaxf(fr * fr, 1e-10f) / (256.f * 256.f);
    hdr[244] = expf(log_step[0]);
  }
}

// --------------------------------------------------------------------------
// Build transposed+padded reference rT[b][HP][WP][256] (bf16, zero halo).
// idx fastest over c -> coalesced stores.
// --------------------------------------------------------------------------
__global__ void cvt_rT_kernel(const float* __restrict__ r, __bf16* __restrict__ rT,
                              size_t n) {
  for (size_t i = (size_t)blockIdx.x * blockDim.x + threadIdx.x; i < n;
       i += (size_t)gridDim.x * blockDim.x) {
    int c  = (int)(i & 255);
    size_t p = i >> 8;
    int xx = (int)(p % WP);
    int yy = (int)((p / WP) % HP);
    int b  = (int)(p / (WP * HP));
    float v = 0.f;
    if (yy >= 4 && yy < 68 && xx >= 4 && xx < 68)
      v = r[(((size_t)b * 256 + c) * 64 + (yy - 4)) * 64 + (xx - 4)];
    rT[i] = (__bf16)v;
  }
}

// --------------------------------------------------------------------------
// Build padded reference rp[b][256][HP][WP] (bf16, zero halo, original order).
// Contiguous reads and writes.
// --------------------------------------------------------------------------
__global__ void cvt_rp_kernel(const float* __restrict__ r, __bf16* __restrict__ rp,
                              size_t n) {
  for (size_t i = (size_t)blockIdx.x * blockDim.x + threadIdx.x; i < n;
       i += (size_t)gridDim.x * blockDim.x) {
    int xx = (int)(i % WP);
    size_t p = i / WP;
    int yy = (int)(p % HP);
    int c  = (int)((p / HP) & 255);
    int b  = (int)(p / (HP * 256));
    float v = 0.f;
    if (yy >= 4 && yy < 68 && xx >= 4 && xx < 68)
      v = r[(((size_t)b * 256 + c) * 64 + (yy - 4)) * 64 + (xx - 4)];
    rp[i] = (__bf16)v;
  }
}

// (b,c,y,x) flat index -> transposed bf16 index [b][y][x][c]
__device__ __forceinline__ size_t t_idx(size_t i) {
  size_t b = i >> 20;
  size_t c = (i >> 12) & 255;
  size_t yx = i & 4095;
  return ((b << 12) + yx) * 256 + c;
}

__global__ void init_kernel(const float* __restrict__ src, float* __restrict__ fdst,
                            __bf16* __restrict__ bT, size_t n) {
  for (size_t i = (size_t)blockIdx.x * blockDim.x + threadIdx.x; i < n;
       i += (size_t)gridDim.x * blockDim.x) {
    float v = src[i];
    fdst[i] = v;
    bT[t_idx(i)] = (__bf16)v;
  }
}

__global__ void cvtT_kernel(const float* __restrict__ src, __bf16* __restrict__ bT,
                            size_t n) {
  for (size_t i = (size_t)blockIdx.x * blockDim.x + threadIdx.x; i < n;
       i += (size_t)gridDim.x * blockDim.x)
    bT[t_idx(i)] = (__bf16)src[i];
}

// --------------------------------------------------------------------------
// scores[b, dy*9+dx, y, x] = sum_c F[b,c,y,x] * R[b,c,y+dy-4,x+dx-4]
// FT: filter bf16 [b][64][64][256]; RT: padded+transposed reference.
// TDM stages the 8KB F tile and double-buffered 16KB R windows into LDS
// (with hardware-inserted 16B/512B bank padding); 8x WMMA bf16 per (dy,wave).
// grid = (4, 64, 16), block = 64 (2 waves; wave = N tile).
// --------------------------------------------------------------------------
__global__ __launch_bounds__(64) void corr_wmma_kernel(
    const __bf16* __restrict__ FT, const __bf16* __restrict__ RT,
    float* __restrict__ scores) {
  __shared__ __bf16 Flds[16][264];
  __shared__ __bf16 Rlds[2][32][264];
  const int x0  = blockIdx.x * 16;
  const int y   = blockIdx.y;
  const int b   = blockIdx.z;
  const int tid = threadIdx.x;
  const int lane = tid & 31;
  const int wv   = tid >> 5;
  const int half = lane >> 4;
  const int m    = lane & 15;
  const int n    = lane & 15;

  const unsigned long long fga =
      (unsigned long long)(size_t)(FT + (((size_t)b * 64 + y) * 64 + x0) * 256);
  if (wv == 0) {
    tdm_load_1d(lds_off32(&Flds[0][0]), fga, 4096);
    unsigned long long rga =
        (unsigned long long)(size_t)(RT + (((size_t)b * HP + y) * WP + x0) * 256);
    tdm_load_1d(lds_off32(&Rlds[0][0][0]), rga, 8192);
    __builtin_amdgcn_s_wait_tensorcnt((short)1);  // F tile complete
  }
  __syncthreads();

  // Preload the 8 A K-chunks (per-lane v16bf per ISA 16-bit A layout).
  v16bf a[8];
  #pragma unroll
  for (int kc = 0; kc < 8; ++kc) {
    const __bf16* p0 = &Flds[m][kc * 32 + half * 8];
    const __bf16* p1 = &Flds[m][kc * 32 + 16 + half * 8];
    #pragma unroll
    for (int i = 0; i < 8; ++i) { a[kc][i] = p0[i]; a[kc][8 + i] = p1[i]; }
  }

  for (int dy = 0; dy < 9; ++dy) {
    const int cur = dy & 1;
    if (wv == 0) {
      if (dy < 8) {
        unsigned long long rga = (unsigned long long)(size_t)(
            RT + (((size_t)b * HP + y + dy + 1) * WP + x0) * 256);
        tdm_load_1d(lds_off32(&Rlds[1 - cur][0][0]), rga, 8192);
        __builtin_amdgcn_s_wait_tensorcnt((short)1);  // buf[cur] done, next in flight
      } else {
        __builtin_amdgcn_s_wait_tensorcnt((short)0);
      }
    }
    __syncthreads();

    // Preload all 8 B chunks, then issue the WMMA chain back-to-back.
    v16bf bb[8];
    #pragma unroll
    for (int kc = 0; kc < 8; ++kc) {
      const __bf16* q0 = &Rlds[cur][wv * 16 + n][kc * 32 + half * 8];
      const __bf16* q1 = &Rlds[cur][wv * 16 + n][kc * 32 + 16 + half * 8];
      #pragma unroll
      for (int i = 0; i < 8; ++i) { bb[kc][i] = q0[i]; bb[kc][8 + i] = q1[i]; }
    }
    v8f acc = {};
    #pragma unroll
    for (int kc = 0; kc < 8; ++kc)
      acc = __builtin_amdgcn_wmma_f32_16x16x32_bf16(false, a[kc], false, bb[kc],
                                                    (short)0, acc, false, false);

    // Band extraction: dxs = wv*16 + N - M, valid 0..8.
    #pragma unroll
    for (int r = 0; r < 8; ++r) {
      int M = r + 8 * half;
      int dxs = wv * 16 + n - M;
      if (dxs >= 0 && dxs <= 8) {
        int d = dy * 9 + dxs;
        scores[(((size_t)b * NDISP + d) * 64 + y) * 64 + x0 + M] = acc[r];
      }
    }
    __syncthreads();  // all reads of Rlds[1-cur] from prior iter are done
  }
}

// --------------------------------------------------------------------------
// mapped = grad_act * (act - v_plus*target)
// --------------------------------------------------------------------------
__global__ void map_kernel(const float* __restrict__ s1, float* __restrict__ s2,
                           const float* __restrict__ hdr, size_t n) {
  for (size_t i = (size_t)blockIdx.x * blockDim.x + threadIdx.x; i < n;
       i += (size_t)gridDim.x * blockDim.x) {
    int d = (int)((i >> 12) % NDISP);
    float s  = s1[i];
    float c0 = hdr[81 + d], c1 = hdr[162 + d], tg = hdr[d];
    float sg = (s > 0.f) ? 1.f : ((s < 0.f) ? -1.f : 0.f);
    s2[i] = (c0 * sg + c1) * (c0 * fabsf(s) + c1 * s - tg);
  }
}

// --------------------------------------------------------------------------
// fgrad[b,c,y,x] = reg*fcur + sum_{dy,dx} mapped[b,d,y,x]*R[b,c,y+dy-4,x+dx-4]
// A = padded reference rows [c][x' window] staged via async-load-to-LDS,
// B = banded matrix built from `mapped`. 16 channel tiles (8/wave), 9 dy.
// grid = (4, 64, 16), block = 64.
// --------------------------------------------------------------------------
__global__ __launch_bounds__(64) void corrT_wmma_kernel(
    const float* __restrict__ mapped, const __bf16* __restrict__ RP,
    const float* __restrict__ fcur, float* __restrict__ fgrad,
    const float* __restrict__ hdr) {
  __shared__ __bf16 Al[256][40];  // [channel][window px], 80B row (16B aligned)
  __shared__ __bf16 Bl[32][24];   // [window px][out px]
  const int x0  = blockIdx.x * 16;
  const int y   = blockIdx.y;
  const int b   = blockIdx.z;
  const int tid = threadIdx.x;
  const int lane = tid & 31;
  const int wv   = tid >> 5;
  const int half = lane >> 4;
  const int m    = lane & 15;
  const int n    = lane & 15;
  const float reg = hdr[243];

  v8f acc[8];
  #pragma unroll
  for (int j = 0; j < 8; ++j) acc[j] = (v8f){};

  for (int dy = 0; dy < 9; ++dy) {
    __syncthreads();
    // Banded B: B[k][nn] = mapped[d = dy*9 + (k-nn)] for 0<=k-nn<=8.
    for (int idx = tid; idx < 512; idx += 64) {
      int k = idx >> 4, nn = idx & 15;
      int dd = k - nn;
      __bf16 v = (__bf16)0.0f;
      if (dd >= 0 && dd <= 8)
        v = (__bf16)mapped[(((size_t)b * NDISP + dy * 9 + dd) * 64 + y) * 64 + x0 + nn];
      Bl[k][nn] = v;
    }
    // A: 256 channel rows x 64B, async DMA (8 rows / wave-instruction).
    {
      int crow = (lane >> 2);          // 0..7
      int chunk = (lane & 3) * 16;     // byte offset within 64B row
      for (int rr = wv * 128; rr < wv * 128 + 128; rr += 8) {
        int c = rr + crow;
        unsigned long long ga = (unsigned long long)(size_t)(
            RP + (((size_t)b * 256 + c) * HP + (y + dy)) * WP + x0) + chunk;
        async_ld_b128(lds_off32(&Al[c][0]) + chunk, ga);
      }
    }
    wait_async0();
    __syncthreads();

    v16bf bb;
    #pragma unroll
    for (int i = 0; i < 8; ++i) {
      bb[i]     = Bl[half * 8 + i][n];
      bb[8 + i] = Bl[16 + half * 8 + i][n];
    }
    #pragma unroll
    for (int j = 0; j < 8; ++j) {
      int ct = wv * 8 + j;
      v16bf a;
      const __bf16* p0 = &Al[ct * 16 + m][half * 8];
      const __bf16* p1 = &Al[ct * 16 + m][16 + half * 8];
      #pragma unroll
      for (int i = 0; i < 8; ++i) { a[i] = p0[i]; a[8 + i] = p1[i]; }
      acc[j] = __builtin_amdgcn_wmma_f32_16x16x32_bf16(false, a, false, bb,
                                                       (short)0, acc[j], false, false);
    }
  }

  #pragma unroll
  for (int j = 0; j < 8; ++j) {
    int ct = wv * 8 + j;
    #pragma unroll
    for (int r = 0; r < 8; ++r) {
      int c = ct * 16 + r + 8 * half;
      size_t idx = (((size_t)b * 256 + c) * 64 + y) * 64 + x0 + n;
      fgrad[idx] = reg * fcur[idx] + acc[j][r];
    }
  }
}

// --------------------------------------------------------------------------
// alpha_num[p] = sum_c fgrad^2
// --------------------------------------------------------------------------
__global__ void anum_kernel(const float* __restrict__ fgrad, float* __restrict__ anum) {
  int p = blockIdx.x * blockDim.x + threadIdx.x;
  int b = p >> 12, yx = p & 4095;
  float s = 0.f;
  for (int c = 0; c < 256; ++c) {
    float v = fgrad[(((size_t)b * 256 + c) << 12) + yx];
    s += v * v;
  }
  anum[p] = s;
}

// --------------------------------------------------------------------------
// ascale[p] = step*num / max(sum_d (grad_act*corr(fgrad))^2 + reg*num, 1e-8)
// --------------------------------------------------------------------------
__global__ void scale_kernel(const float* __restrict__ s1, const float* __restrict__ s2,
                             const float* __restrict__ anum, float* __restrict__ ascale,
                             const float* __restrict__ hdr) {
  int p = blockIdx.x * blockDim.x + threadIdx.x;
  int b = p >> 12, yx = p & 4095;
  float den = 0.f;
  for (int d = 0; d < NDISP; ++d) {
    size_t idx = (((size_t)b * NDISP + d) << 12) + yx;
    float s = s1[idx];
    float sg = (s > 0.f) ? 1.f : ((s < 0.f) ? -1.f : 0.f);
    float t = (hdr[81 + d] * sg + hdr[162 + d]) * s2[idx];
    den += t * t;
  }
  float num = anum[p];
  den = fmaxf(den + hdr[243] * num, 1e-8f);
  ascale[p] = hdr[244] * num / den;
}

// --------------------------------------------------------------------------
// fcur -= ascale*fgrad ; refresh transposed bf16 copy for next iteration
// --------------------------------------------------------------------------
__global__ void update_kernel(const float* __restrict__ fcur, const float* __restrict__ fgrad,
                              const float* __restrict__ ascale, float* __restrict__ fout,
                              __bf16* __restrict__ bT, size_t nf) {
  for (size_t i = (size_t)blockIdx.x * blockDim.x + threadIdx.x; i < nf;
       i += (size_t)gridDim.x * blockDim.x) {
    int p = (int)(((i >> 20) << 12) | (i & 4095));
    float v = fcur[i] - ascale[p] * fgrad[i];
    fout[i] = v;
    bT[t_idx(i)] = (__bf16)v;
  }
}

extern "C" void kernel_launch(void* const* d_in, const int* in_sizes, int n_in,
                              void* d_out, int out_size, void* d_ws, size_t ws_size,
                              hipStream_t stream) {
  const float* f_in      = (const float*)d_in[0];
  const float* r_in      = (const float*)d_in[1];
  const float* log_step  = (const float*)d_in[2];
  const float* freg      = (const float*)d_in[3];
  const float* label_w   = (const float*)d_in[4];
  const float* spatial_w = (const float*)d_in[5];
  const float* mask_w    = (const float*)d_in[6];
  float* out = (float*)d_out;

  const size_t NF   = (size_t)16 * 256 * 64 * 64;    // 16.7M
  const size_t NS   = (size_t)16 * NDISP * 64 * 64;  // 5.3M
  const size_t NP   = (size_t)16 * 64 * 64;
  const size_t NPAD = (size_t)16 * HP * WP * 256;    // 23.6M

  char* ws = (char*)d_ws;
  size_t off = 0;
  float*  hdr    = (float*)(ws + off);  off += 4096;
  float*  fcur   = (float*)(ws + off);  off += NF * 4;
  float*  fgrad  = (float*)(ws + off);  off += NF * 4;
  __bf16* rT     = (__bf16*)(ws + off); off += NPAD * 2;
  __bf16* rp     = (__bf16*)(ws + off); off += NPAD * 2;
  __bf16* xbT    = (__bf16*)(ws + off); off += NF * 2;
  float*  S1     = (float*)(ws + off);  off += NS * 4;
  float*  S2     = (float*)(ws + off);  off += NS * 4;
  float*  anum   = (float*)(ws + off);  off += NP * 4;
  float*  ascale = (float*)(ws + off);  off += NP * 4;
  (void)ws_size; (void)in_sizes; (void)n_in; (void)out_size;

  prep_kernel<<<1, 128, 0, stream>>>(log_step, freg, label_w, spatial_w, mask_w, hdr);
  cvt_rT_kernel<<<4096, 256, 0, stream>>>(r_in, rT, NPAD);
  cvt_rp_kernel<<<4096, 256, 0, stream>>>(r_in, rp, NPAD);
  init_kernel<<<2048, 256, 0, stream>>>(f_in, fcur, xbT, NF);

  dim3 cg(4, 64, 16);
  for (int it = 0; it < 3; ++it) {
    corr_wmma_kernel<<<cg, 64, 0, stream>>>(xbT, rT, S1);
    map_kernel<<<2048, 256, 0, stream>>>(S1, S2, hdr, NS);
    corrT_wmma_kernel<<<cg, 64, 0, stream>>>(S2, rp, fcur, fgrad, hdr);
    cvtT_kernel<<<2048, 256, 0, stream>>>(fgrad, xbT, NF);
    corr_wmma_kernel<<<cg, 64, 0, stream>>>(xbT, rT, S2);
    anum_kernel<<<256, 256, 0, stream>>>(fgrad, anum);
    scale_kernel<<<256, 256, 0, stream>>>(S1, S2, anum, ascale, hdr);
    float* fo = (it == 2) ? out : fcur;
    update_kernel<<<4096, 256, 0, stream>>>(fcur, fgrad, ascale, fo, xbT, NF);
  }
}
